// EquivariantTransformer_7275674599757
// MI455X (gfx1250) — matrix operand order repaired and checked by reading, not compiled, not run on hardware
//
#include <hip/hip_runtime.h>
#include <hip/hip_bf16.h>

typedef __bf16 v16bf __attribute__((ext_vector_type(16)));
typedef float  v8f   __attribute__((ext_vector_type(8)));

struct __align__(16) US8 { unsigned short s[8]; };   // POD 16-byte chunk -> b128 ld/st

union Frag { v16bf v; unsigned short u[16]; US8 h[2]; };

__device__ __forceinline__ unsigned short f2bf(float x) {
    unsigned int u = __float_as_uint(x);
    u += 0x7FFFu + ((u >> 16) & 1u);          // round-to-nearest-even
    return (unsigned short)(u >> 16);
}

__device__ __forceinline__ v8f wmma_bf16(v16bf a, v16bf b, v8f c) {
    return __builtin_amdgcn_wmma_f32_16x16x32_bf16(
        false, a, false, b, (short)0, c, false, false);
}

#define NN   4096
#define DD   128
#define HH   4
#define HDIM 32
#define LL   4
#define DCC  64
#define ASPLIT 8                      // waves per n-tile in attention

// ---------------- small helper kernels ----------------

__global__ void et_init_out(float* out) { out[0] = 0.0f; }

__global__ void et_pos_mean(const float* __restrict__ pos, float* __restrict__ mean) {
    __shared__ float red[256 * 3];
    int t = threadIdx.x;
    float s0 = 0.f, s1 = 0.f, s2 = 0.f;
    for (int n = t; n < NN; n += 256) {
        s0 += pos[n * 3 + 0]; s1 += pos[n * 3 + 1]; s2 += pos[n * 3 + 2];
    }
    red[t] = s0; red[t + 256] = s1; red[t + 512] = s2;
    __syncthreads();
    for (int s = 128; s > 0; s >>= 1) {
        if (t < s) { red[t] += red[t + s]; red[t + 256] += red[t + 256 + s]; red[t + 512] += red[t + 512 + s]; }
        __syncthreads();
    }
    if (t == 0) { mean[0] = red[0] / NN; mean[1] = red[256] / NN; mean[2] = red[512] / NN; }
}

__global__ void et_prep(const float* __restrict__ pos, const float* __restrict__ mean,
                        float* __restrict__ posw, float* __restrict__ relm) {
    int n = blockIdx.x * blockDim.x + threadIdx.x;
    if (n < NN) {
        for (int c = 0; c < 3; ++c) {
            float p = pos[n * 3 + c];
            posw[n * 3 + c] = p;
            relm[n * 3 + c] = p - mean[c];
        }
    }
}

__global__ void et_embed(const float* __restrict__ na, const float* __restrict__ W,
                         const float* __restrict__ b, float* __restrict__ h,
                         unsigned short* __restrict__ hb) {
    int idx = blockIdx.x * blockDim.x + threadIdx.x;
    if (idx >= NN * DD) return;
    int n = idx >> 7, d = idx & 127;
    float acc = b[d];
    #pragma unroll
    for (int k = 0; k < 16; ++k) acc += na[n * 16 + k] * W[k * DD + d];
    h[idx] = acc;
    hb[idx] = f2bf(acc);
}

// Pack a [L,128,128] f32 weight tensor into per-(n-tile,k-block,lane) B-fragment order,
// so each lane's WMMA B operand is two contiguous b128 loads.
// Bp[l][tn(8)][kblk(4)][lane(32)][e(16)] = bf16( W[l][kblk*32 + (e&7) + 8*(lane>>4) + 16*(e>>3)][tn*16 + (lane&15)] )
__global__ void et_pack_w(const float* __restrict__ W, unsigned short* __restrict__ Bp) {
    int idx = blockIdx.x * blockDim.x + threadIdx.x;
    if (idx >= LL * DD * DD) return;
    int e    = idx & 15;
    int lane = (idx >> 4) & 31;
    int kblk = (idx >> 9) & 3;
    int tn   = (idx >> 11) & 7;
    int l    = idx >> 14;
    int row = lane & 15, hi = lane >> 4;
    int K = kblk * 32 + (e & 7) + (hi << 3) + ((e >> 3) << 4);
    int N = (tn << 4) + row;
    Bp[idx] = f2bf(W[l * DD * DD + K * DD + N]);
}

// ---------------- WMMA GEMM: C[M,128] = Abf[M,128] @ W + bias ----------------
// Each wave computes a 16x32 slab (two n-tiles) so the A fragment is reused across
// two WMMAs per k-step. MODE 0: f32 row-major; 1: bf16 row-major; 2: bf16 transposed.
template<int MODE>
__global__ void et_gemm(const unsigned short* __restrict__ A,   // [M,128] row-major bf16
                        const unsigned short* __restrict__ Bp,  // packed fragments (one layer)
                        const float* __restrict__ bias,
                        void* __restrict__ Cout, int M) {
    int wave = blockIdx.x * (blockDim.x >> 5) + (threadIdx.x >> 5);
    int tp = wave & 3;                  // n-tile pair: tiles 2tp, 2tp+1
    int m0 = (wave >> 2) << 4;
    int lane = threadIdx.x & 31, row = lane & 15, hi = lane >> 4;

    v8f acc0 = {0.f, 0.f, 0.f, 0.f, 0.f, 0.f, 0.f, 0.f};
    v8f acc1 = {0.f, 0.f, 0.f, 0.f, 0.f, 0.f, 0.f, 0.f};
    const unsigned short* arow = A + (m0 + row) * DD;
    #pragma unroll
    for (int kb = 0; kb < 4; ++kb) {
        Frag a, b0, b1;
        a.h[0] = *(const US8*)(arow + kb * 32 + (hi << 3));        // K = kb*32+8hi .. +7
        a.h[1] = *(const US8*)(arow + kb * 32 + 16 + (hi << 3));   // K = +16 ..
        const US8* bp0 = (const US8*)(Bp + (((2 * tp + 0) * 4 + kb) * 32 + lane) * 16);
        const US8* bp1 = (const US8*)(Bp + (((2 * tp + 1) * 4 + kb) * 32 + lane) * 16);
        b0.h[0] = bp0[0]; b0.h[1] = bp0[1];
        b1.h[0] = bp1[0]; b1.h[1] = bp1[1];
        acc0 = wmma_bf16(a.v, b0.v, acc0);
        acc1 = wmma_bf16(a.v, b1.v, acc1);
    }
    #pragma unroll
    for (int t = 0; t < 2; ++t) {
        v8f acc = t ? acc1 : acc0;
        int n0 = (2 * tp + t) << 4;
        float bs = bias[n0 + row];
        if (MODE == 0) {
            float* C = (float*)Cout;
            #pragma unroll
            for (int r = 0; r < 8; ++r)
                C[(m0 + r + (hi << 3)) * DD + n0 + row] = acc[r] + bs;
        } else if (MODE == 1) {
            unsigned short* C = (unsigned short*)Cout;
            #pragma unroll
            for (int r = 0; r < 8; ++r)
                C[(m0 + r + (hi << 3)) * DD + n0 + row] = f2bf(acc[r] + bs);
        } else {
            unsigned short* C = (unsigned short*)Cout;   // [128, M] transposed
            US8 tv;
            #pragma unroll
            for (int r = 0; r < 8; ++r) tv.s[r] = f2bf(acc[r] + bs);
            *(US8*)(C + (n0 + row) * M + m0 + (hi << 3)) = tv;     // contiguous b128 store
        }
    }
}

// ---------------- fused attention (softmax over HEAD axis) ----------------
// Scores computed transposed (S^T = K_tile @ Q^T): softmaxed D-fragment == A-fragment of
// attn@V, so no cross-lane repack. 8 waves/block split the m-range; Q tile staged in LDS;
// LDS float atomics combine partial output tiles; next-block K rows prefetched.
__global__ void __launch_bounds__(32 * ASPLIT, 1)
et_attention(const unsigned short* __restrict__ qb,
             const unsigned short* __restrict__ kb,
             const unsigned short* __restrict__ vt,   // [128, N] transposed
             unsigned short* __restrict__ attnb) {
    __shared__ float lds[16 * DD];                    // 8 KB partial-sum tile
    __shared__ unsigned short qs[16 * DD];            // 4 KB staged q tile
    int n0 = blockIdx.x << 4;
    int wslot = threadIdx.x >> 5;
    int lane = threadIdx.x & 31, row = lane & 15, hi = lane >> 4;
    const float scale = 0.17677669529663687f;         // 1/sqrt(32)

    // stage q tile + zero the reduce buffer, one barrier
    {
        int i = threadIdx.x;                          // 256 threads x 8 ushorts = 16x128
        *(US8*)(qs + i * 8) = *(const US8*)(qb + n0 * DD + i * 8);
        lds[i] = 0.f; lds[i + 256] = 0.f; lds[i + 512] = 0.f; lds[i + 768] = 0.f;
        lds[i + 1024] = 0.f; lds[i + 1280] = 0.f; lds[i + 1536] = 0.f; lds[i + 1792] = 0.f;
    }
    __syncthreads();

    v8f acc[HH][2];
    #pragma unroll
    for (int h = 0; h < HH; ++h)
        #pragma unroll
        for (int x = 0; x < 2; ++x)
            acc[h][x] = (v8f){0.f, 0.f, 0.f, 0.f, 0.f, 0.f, 0.f, 0.f};

    const int mspan = NN / ASPLIT;
    const int mlo = wslot * mspan;
    for (int m0 = mlo; m0 < mlo + mspan; m0 += 32) {
        // prefetch next m-block's K rows (sequential stream, one block ahead)
        __builtin_prefetch(kb + (m0 + 32 + row) * DD, 0, 1);
        __builtin_prefetch(kb + (m0 + 48 + row) * DD, 0, 1);
        v8f s0[HH], s1[HH];
        #pragma unroll
        for (int h = 0; h < HH; ++h) {
            Frag qf;                                  // re-read from LDS (2x ds_load_b128)
            const US8* qp = (const US8*)(qs + row * DD + h * HDIM + (hi << 3));
            qf.h[0] = qp[0]; qf.h[1] = qp[2];
            Frag ka0, ka1;
            const unsigned short* k0p = kb + (m0 + row) * DD + h * HDIM;
            const unsigned short* k1p = kb + (m0 + 16 + row) * DD + h * HDIM;
            ka0.h[0] = *(const US8*)(k0p + (hi << 3));
            ka0.h[1] = *(const US8*)(k0p + 16 + (hi << 3));
            ka1.h[0] = *(const US8*)(k1p + (hi << 3));
            ka1.h[1] = *(const US8*)(k1p + 16 + (hi << 3));
            v8f z = {0.f, 0.f, 0.f, 0.f, 0.f, 0.f, 0.f, 0.f};
            s0[h] = wmma_bf16(ka0.v, qf.v, z);
            s1[h] = wmma_bf16(ka1.v, qf.v, z);
        }
        Frag af[HH];
        #pragma unroll
        for (int r = 0; r < 8; ++r) {
            {
                float x0 = s0[0][r] * scale, x1 = s0[1][r] * scale,
                      x2 = s0[2][r] * scale, x3 = s0[3][r] * scale;
                float mx = fmaxf(fmaxf(x0, x1), fmaxf(x2, x3));
                float e0 = __expf(x0 - mx), e1 = __expf(x1 - mx),
                      e2 = __expf(x2 - mx), e3 = __expf(x3 - mx);
                float inv = 1.0f / (e0 + e1 + e2 + e3);
                af[0].u[r] = f2bf(e0 * inv); af[1].u[r] = f2bf(e1 * inv);
                af[2].u[r] = f2bf(e2 * inv); af[3].u[r] = f2bf(e3 * inv);
            }
            {
                float x0 = s1[0][r] * scale, x1 = s1[1][r] * scale,
                      x2 = s1[2][r] * scale, x3 = s1[3][r] * scale;
                float mx = fmaxf(fmaxf(x0, x1), fmaxf(x2, x3));
                float e0 = __expf(x0 - mx), e1 = __expf(x1 - mx),
                      e2 = __expf(x2 - mx), e3 = __expf(x3 - mx);
                float inv = 1.0f / (e0 + e1 + e2 + e3);
                af[0].u[r + 8] = f2bf(e0 * inv); af[1].u[r + 8] = f2bf(e1 * inv);
                af[2].u[r + 8] = f2bf(e2 * inv); af[3].u[r + 8] = f2bf(e3 * inv);
            }
        }
        #pragma unroll
        for (int h = 0; h < HH; ++h)
            #pragma unroll
            for (int x = 0; x < 2; ++x) {
                Frag vf;   // V from transposed store: two 16B chunks per lane (k 8hi.., 16+8hi..)
                const US8* vp = (const US8*)(vt + (h * HDIM + x * 16 + row) * NN + m0 + (hi << 3));
                vf.h[0] = vp[0]; vf.h[1] = vp[2];
                acc[h][x] = wmma_bf16(af[h].v, vf.v, acc[h][x]);
            }
    }
    // combine partial tiles across the 8 waves via LDS float atomics (ds_add_f32)
    #pragma unroll
    for (int h = 0; h < HH; ++h)
        #pragma unroll
        for (int x = 0; x < 2; ++x)
            #pragma unroll
            for (int r = 0; r < 8; ++r)
                atomicAdd(&lds[(r + (hi << 3)) * DD + h * HDIM + x * 16 + row], acc[h][x][r]);
    __syncthreads();
    // one b128 bf16 store per thread (256 threads x 8 elems = 16x128 tile)
    {
        int i = threadIdx.x;
        int nl = i >> 4;
        int d0 = (i & 15) << 3;
        US8 t;
        #pragma unroll
        for (int j = 0; j < 8; ++j) t.s[j] = f2bf(lds[nl * DD + d0 + j]);
        *(US8*)(attnb + (n0 + nl) * DD + d0) = t;
    }
}

// ---------------- residual + layernorm ----------------

__global__ void et_layernorm(float* __restrict__ h, const float* __restrict__ o,
                             const float* __restrict__ g, const float* __restrict__ b,
                             unsigned short* __restrict__ hb) {
    __shared__ float red[DD];
    int n = blockIdx.x, d = threadIdx.x;
    float x = h[n * DD + d] + o[n * DD + d];
    red[d] = x; __syncthreads();
    for (int s = 64; s > 0; s >>= 1) { if (d < s) red[d] += red[d + s]; __syncthreads(); }
    float mu = red[0] * (1.0f / DD);
    __syncthreads();
    float dx = x - mu;
    red[d] = dx * dx; __syncthreads();
    for (int s = 64; s > 0; s >>= 1) { if (d < s) red[d] += red[d + s]; __syncthreads(); }
    float var = red[0] * (1.0f / DD);
    float y = dx * rsqrtf(var + 1e-5f) * g[d] + b[d];
    h[n * DD + d] = y;
    hb[n * DD + d] = f2bf(y);
}

// ---------------- coordinate MLP + pos update ----------------

__global__ void et_coord(const float* __restrict__ h, const float* __restrict__ Wc1,
                         const float* __restrict__ bc1, const float* __restrict__ Wc2,
                         const float* __restrict__ bc2, const float* __restrict__ relm,
                         float* __restrict__ posw) {
    __shared__ float red[DCC];
    int n = blockIdx.x, j = threadIdx.x;
    float acc = bc1[j];
    for (int k = 0; k < DD; ++k) acc += h[n * DD + k] * Wc1[k * DCC + j];
    float s = acc / (1.0f + __expf(-acc));               // silu
    red[j] = s * Wc2[j];
    __syncthreads();
    for (int t = 32; t > 0; t >>= 1) { if (j < t) red[j] += red[j + t]; __syncthreads(); }
    if (j == 0) {
        float cw = red[0] + bc2[0];
        posw[n * 3 + 0] += cw * relm[n * 3 + 0];
        posw[n * 3 + 1] += cw * relm[n * 3 + 1];
        posw[n * 3 + 2] += cw * relm[n * 3 + 2];
    }
}

// ---------------- final energy + pos output ----------------

__global__ void et_energy(const float* __restrict__ h, const float* __restrict__ We,
                          const float* __restrict__ be, float* __restrict__ out) {
    __shared__ float red[256];
    int t = threadIdx.x, gid = blockIdx.x * blockDim.x + t;
    float p = 0.f;
    for (int i = gid; i < NN * DD; i += gridDim.x * blockDim.x)
        p += h[i] * We[i & 127];
    red[t] = p; __syncthreads();
    for (int s = 128; s > 0; s >>= 1) { if (t < s) red[t] += red[t + s]; __syncthreads(); }
    if (t == 0) {
        float v = red[0];
        if (blockIdx.x == 0) v += (float)NN * be[0];
        atomicAdd(out, v);
    }
}

__global__ void et_poscopy(const float* __restrict__ posw, float* __restrict__ out) {
    int i = blockIdx.x * blockDim.x + threadIdx.x;
    if (i < NN * 3) out[1 + i] = posw[i];
}

// ---------------- host launcher ----------------

extern "C" void kernel_launch(void* const* d_in, const int* in_sizes, int n_in,
                              void* d_out, int out_size, void* d_ws, size_t ws_size,
                              hipStream_t stream) {
    const float* node_attr = (const float*)d_in[0];
    const float* pos    = (const float*)d_in[1];
    const float* emb_W  = (const float*)d_in[2];
    const float* emb_b  = (const float*)d_in[3];
    const float* Wq     = (const float*)d_in[4];
    const float* bq     = (const float*)d_in[5];
    const float* Wk     = (const float*)d_in[6];
    const float* bk     = (const float*)d_in[7];
    const float* Wv     = (const float*)d_in[8];
    const float* bv     = (const float*)d_in[9];
    const float* Wo     = (const float*)d_in[10];
    const float* bo     = (const float*)d_in[11];
    const float* Wc1    = (const float*)d_in[12];
    const float* bc1    = (const float*)d_in[13];
    const float* Wc2    = (const float*)d_in[14];
    const float* bc2    = (const float*)d_in[15];
    const float* ln_g   = (const float*)d_in[16];
    const float* ln_b   = (const float*)d_in[17];
    const float* We     = (const float*)d_in[18];
    const float* be     = (const float*)d_in[19];
    float* out = (float*)d_out;

    char* w = (char*)d_ws;
    const size_t MB = 1u << 20;
    float*          h     = (float*)(w + 0 * MB);            // 2 MB
    float*          o     = (float*)(w + 2 * MB);            // 2 MB
    unsigned short* hb    = (unsigned short*)(w + 4 * MB);   // 1 MB each below
    unsigned short* qb    = (unsigned short*)(w + 5 * MB);
    unsigned short* kbuf  = (unsigned short*)(w + 6 * MB);
    unsigned short* vt    = (unsigned short*)(w + 7 * MB);   // transposed [128, N]
    unsigned short* attnb = (unsigned short*)(w + 8 * MB);
    unsigned short* WqB   = (unsigned short*)(w + 9 * MB);               // 128 KB each (packed)
    unsigned short* WkB   = (unsigned short*)(w + 9 * MB + 128 * 1024);
    unsigned short* WvB   = (unsigned short*)(w + 9 * MB + 256 * 1024);
    unsigned short* WoB   = (unsigned short*)(w + 9 * MB + 384 * 1024);
    float*          posw  = (float*)(w + 9 * MB + 512 * 1024);
    float*          relm  = (float*)(w + 9 * MB + 576 * 1024);
    float*          meanp = (float*)(w + 9 * MB + 640 * 1024);

    et_init_out<<<1, 1, 0, stream>>>(out);
    et_pos_mean<<<1, 256, 0, stream>>>(pos, meanp);
    et_prep<<<(NN + 255) / 256, 256, 0, stream>>>(pos, meanp, posw, relm);
    et_embed<<<(NN * DD) / 256, 256, 0, stream>>>(node_attr, emb_W, emb_b, h, hb);

    const int WCNT = LL * DD * DD;
    et_pack_w<<<WCNT / 256, 256, 0, stream>>>(Wq, WqB);
    et_pack_w<<<WCNT / 256, 256, 0, stream>>>(Wk, WkB);
    et_pack_w<<<WCNT / 256, 256, 0, stream>>>(Wv, WvB);
    et_pack_w<<<WCNT / 256, 256, 0, stream>>>(Wo, WoB);

    dim3 gGemm(128), bGemm(256);     // 1024 waves = (4096/16) * 4 n-tile-pairs
    dim3 gAttn(NN / 16), bAttn(32 * ASPLIT);

    for (int l = 0; l < LL; ++l) {
        et_gemm<1><<<gGemm, bGemm, 0, stream>>>(hb, WqB + l * DD * DD, bq + l * DD, qb, NN);
        et_gemm<1><<<gGemm, bGemm, 0, stream>>>(hb, WkB + l * DD * DD, bk + l * DD, kbuf, NN);
        et_gemm<2><<<gGemm, bGemm, 0, stream>>>(hb, WvB + l * DD * DD, bv + l * DD, vt, NN);
        et_attention<<<gAttn, bAttn, 0, stream>>>(qb, kbuf, vt, attnb);
        et_gemm<0><<<gGemm, bGemm, 0, stream>>>(attnb, WoB + l * DD * DD, bo + l * DD, o, NN);
        et_layernorm<<<NN, DD, 0, stream>>>(h, o, ln_g + l * DD, ln_b + l * DD, hb);
        et_coord<<<NN, DCC, 0, stream>>>(h, Wc1 + l * DD * DCC, bc1 + l * DCC,
                                         Wc2 + l * DCC, bc2 + l, relm, posw);
    }

    et_energy<<<256, 256, 0, stream>>>(h, We, be, out);
    et_poscopy<<<(NN * 3 + 255) / 256, 256, 0, stream>>>(posw, out);
}
